// MSA_Conv_2d_13786845020947
// MI455X (gfx1250) — matrix operand order repaired and checked by reading, compiled, not run on hardware
//
#include <hip/hip_runtime.h>
#include <hip/hip_bf16.h>

typedef unsigned short u16;
typedef u16   v8u  __attribute__((ext_vector_type(8)));
typedef u16   v16u __attribute__((ext_vector_type(16)));
typedef float v8f  __attribute__((ext_vector_type(8)));
typedef __bf16 v16bf __attribute__((ext_vector_type(16)));
typedef unsigned int v4u32 __attribute__((ext_vector_type(4)));
typedef int v4i32 __attribute__((ext_vector_type(4)));
typedef int v8i32 __attribute__((ext_vector_type(8)));

#define BATCH 4
#define CCH   64
#define HH    256
#define WW    256
#define PIX   (HH*WW)   /* 65536 = 1<<16 */
#define OC3   192       /* 3*C  */
#define HP    258       /* padded rows  (1 top, 1 bottom) */
#define WP    260       /* padded cols  (2 left, 2 right) */

#if defined(__gfx1250__) && __has_builtin(__builtin_amdgcn_tensor_load_to_lds) && \
    __has_builtin(__builtin_amdgcn_s_wait_tensorcnt)
#define USE_TDM 1
#else
#define USE_TDM 0
#endif

// ---- bf16 helpers (RNE) --------------------------------------------------
__device__ __forceinline__ u16 f2bf(float f) {
  unsigned u = __float_as_uint(f);
  u += 0x7FFFu + ((u >> 16) & 1u);
  return (u16)(u >> 16);
}
// Two contiguous 16B loads -> one 32B WMMA operand vector
__device__ __forceinline__ v16bf mk16(const u16* p0, const u16* p1) {
  v8u lo = *(const v8u*)p0;
  v8u hi = *(const v8u*)p1;
  v16u v = __builtin_shufflevector(lo, hi, 0,1,2,3,4,5,6,7,8,9,10,11,12,13,14,15);
  return __builtin_bit_cast(v16bf, v);
}
__device__ __forceinline__ v8f wmma_bf16(v16bf a, v16bf b, v8f c) {
  return __builtin_amdgcn_wmma_f32_16x16x32_bf16(false, a, false, b, (short)0, c,
                                                 false, false);
}

#if USE_TDM
// ---- Tensor Data Mover descriptor helpers (D# group0/group1) -------------
__device__ __forceinline__ v4u32 tdm_g0(unsigned lds_byte_addr,
                                        unsigned long long global_addr) {
  v4u32 g;
  g[0] = 1u;                                        // count = 1 (valid D#)
  g[1] = lds_byte_addr;                             // lds_addr [63:32]
  g[2] = (unsigned)global_addr;                     // global_addr [95:64]
  g[3] = (unsigned)(global_addr >> 32) | 0x80000000u; // addr hi + type=2 [127:126]
  return g;
}
// data_size = 8B units; 2-D tile: tile0 x tile1, line stride stride0
__device__ __forceinline__ v8i32 tdm_g1(unsigned td0, unsigned td1,
                                        unsigned tile0, unsigned tile1,
                                        unsigned stride0) {
  v8i32 g;
  g[0] = 0x00030000;                                // data_size=3 (8 bytes)
  g[1] = (int)(td0 << 16);                          // tensor_dim0 [79:48]
  g[2] = (int)((td0 >> 16) | (td1 << 16));          // tensor_dim1 [111:80]
  g[3] = (int)((td1 >> 16) | (tile0 << 16));        // tile_dim0  [127:112]
  g[4] = (int)(tile1 & 0xFFFFu);                    // tile_dim1  [143:128]
  g[5] = (int)stride0;                              // tensor_dim0_stride lo32
  g[6] = 0;
  g[7] = 0;
  return g;
}
__device__ __forceinline__ void tdm_load(v4u32 g0, v8i32 g1) {
  v4i32 z4 = {0, 0, 0, 0};
#if __clang_major__ >= 23
  v8i32 z8 = {0, 0, 0, 0, 0, 0, 0, 0};
  __builtin_amdgcn_tensor_load_to_lds(g0, g1, z4, z4, z8, 0);
#else
  __builtin_amdgcn_tensor_load_to_lds(g0, g1, z4, z4, 0);
#endif
}
#endif  // USE_TDM

// ---- kernel 0: repack weights to bf16 ------------------------------------
// wm  : w_mlp [o][c][ky][kx] fp32 -> [t][o][c] bf16   (36864 elems)
// wqb : w_qkv [o][c]          fp32 -> [o][c]   bf16   (12288 elems)
__global__ __launch_bounds__(256) void conv_w_kernel(const float* __restrict__ wmlp,
                                                     const float* __restrict__ wqkv,
                                                     u16* __restrict__ wm,
                                                     u16* __restrict__ wqb) {
  int i = blockIdx.x * 256 + threadIdx.x;      // i < 49152
  if (i < 36864) {
    int c = i & 63;
    int o = (i >> 6) & 63;
    int t = i >> 12;
    wm[i] = f2bf(wmlp[(o * 64 + c) * 9 + t]);
  } else {
    int j = i - 36864;                         // j < 12288
    wqb[j] = f2bf(wqkv[j]);
  }
}

// ---- kernel 1: qkv = W_qkv @ x + b  (bf16 WMMA, fp32 accum) --------------
// Output pixel-major: qkvp[b][p][o] (o = 0..63 q, 64..127 k, 128..191 v)
__global__ __launch_bounds__(256) void qkv_kernel(const float* __restrict__ x,
                                                  const u16* __restrict__ wqb,
                                                  const float* __restrict__ bq_g,
                                                  u16* __restrict__ qkvp) {
  __shared__ u16 wq[OC3][CCH];       // weights, c-contiguous (byte copy of wqb)
  __shared__ u16 xs[128][CCH];       // activations, transposed: [pixel][channel]
  __shared__ float bq[OC3];

  int tid = threadIdx.x;
  int blk = blockIdx.x;              // 2048 blocks = B * (PIX/128)
  int b   = blk >> 9;
  int p0  = (blk & 511) << 7;

#if USE_TDM
  if (tid < 32) {
    // 1-D TDM: 192*64 bf16 = 24576 B = 3072 x 8B units, single line
    unsigned ldsa = (unsigned)(size_t)(void*)&wq[0][0];
    tdm_load(tdm_g0(ldsa, (unsigned long long)(size_t)(const void*)wqb),
             tdm_g1(3072u, 1u, 3072u, 1u, 3072u));
  }
#else
  {
    u16* wqf = &wq[0][0];
    for (int i = tid; i < OC3 * CCH; i += 256) wqf[i] = wqb[i];
  }
#endif
  if (tid < OC3) bq[tid] = bq_g[tid];
  for (int i = tid; i < CCH * 128; i += 256) {
    int c = i >> 7, px = i & 127;    // px inner -> coalesced global reads
    xs[px][c] = f2bf(x[((b * CCH + c) << 16) + p0 + px]);
  }
#if USE_TDM
  if (tid < 32) __builtin_amdgcn_s_wait_tensorcnt(0);
#endif
  __syncthreads();

  int lane  = tid & 31;
  int wave  = tid >> 5;
  int n     = lane & 15;
  int cb    = (lane < 16) ? 0 : 16;  // B-matrix K half
  int klo   = (lane < 16) ? 0 : 8;   // unused; A indexes K by cb, like B
  int pxl   = wave * 16 + n;
  int mbase = (lane < 16) ? 0 : 8;
  int pout  = p0 + wave * 16 + n;

  v16bf B0 = mk16(&xs[pxl][cb],      &xs[pxl][cb + 8]);        // K = 0..31
  v16bf B1 = mk16(&xs[pxl][32 + cb], &xs[pxl][32 + cb + 8]);   // K = 32..63

  u16* orow = qkvp + (size_t)((b << 16) + pout) * OC3;
#pragma unroll
  for (int mt = 0; mt < 12; ++mt) {
    int arow = mt * 16 + n;
    v16bf A0 = mk16(&wq[arow][cb],      &wq[arow][cb + 8]);
    v16bf A1 = mk16(&wq[arow][32 + cb], &wq[arow][32 + cb + 8]);
    v8f acc = {};
    acc = wmma_bf16(A0, B0, acc);
    acc = wmma_bf16(A1, B1, acc);
    v8u pack;
#pragma unroll
    for (int r = 0; r < 8; ++r)
      pack[r] = f2bf(acc[r] + bq[mt * 16 + mbase + r]);
    *(v8u*)(orow + mt * 16 + mbase) = pack;    // one b128 store per M-tile
  }
}

// ---- kernel 2: neighborhood attention + residual (fp32 VALU) -------------
// qkvp pixel-major; output pixel-major with zero halo: ypad[b][h+1][w+2][c].
__global__ __launch_bounds__(256) void attn_kernel(const u16* __restrict__ qkvp,
                                                   const float* __restrict__ x,
                                                   u16* __restrict__ ypad) {
  int g = blockIdx.x * 256 + threadIdx.x;   // 1024 blocks = B*PIX/256
  int b = g >> 16;
  int p = g & 65535;
  int h = p >> 8, w = p & 255;

  const u16* row = qkvp + (size_t)((b << 16) + p) * OC3;

  float dots[9];
  int off[9];
  bool ok[9];
#pragma unroll
  for (int t = 0; t < 9; ++t) {
    int dh = t / 3 - 1, dw = t % 3 - 1;
    ok[t]   = ((unsigned)(h + dh) < HH) && ((unsigned)(w + dw) < WW);
    off[t]  = (dh * WW + dw) * OC3;
    dots[t] = 0.f;                 // zero-padded taps keep dot = 0 (matches ref)
  }

  // q: 8 contiguous b128 loads, unpack bf16 pairs with shift/mask
  float qf[CCH];
#pragma unroll
  for (int j = 0; j < 8; ++j) {
    v4u32 qc = *(const v4u32*)(row + j * 8);
#pragma unroll
    for (int e = 0; e < 4; ++e) {
      unsigned u = qc[e];
      qf[j * 8 + 2 * e]     = __uint_as_float(u << 16);
      qf[j * 8 + 2 * e + 1] = __uint_as_float(u & 0xFFFF0000u);
    }
  }
#pragma unroll
  for (int t = 0; t < 9; ++t) {
    if (!ok[t]) continue;
    const u16* kr = row + off[t] + CCH;     // k section
    float d = 0.f;
#pragma unroll
    for (int j = 0; j < 8; ++j) {
      v4u32 kc = *(const v4u32*)(kr + j * 8);
#pragma unroll
      for (int e = 0; e < 4; ++e) {
        unsigned u = kc[e];
        d += qf[j * 8 + 2 * e]     * __uint_as_float(u << 16);
        d += qf[j * 8 + 2 * e + 1] * __uint_as_float(u & 0xFFFF0000u);
      }
    }
    dots[t] = d;
  }

  float m = -1e30f;
#pragma unroll
  for (int t = 0; t < 9; ++t) { dots[t] *= 0.125f; m = fmaxf(m, dots[t]); }
  float s = 0.f;
#pragma unroll
  for (int t = 0; t < 9; ++t) { dots[t] = __expf(dots[t] - m); s += dots[t]; }
  float inv = 1.f / s;
#pragma unroll
  for (int t = 0; t < 9; ++t) dots[t] *= inv;

  int ybase = (((b * HP) + h + 1) * WP + (w + 2)) * CCH;  // pixel-major, haloed
#pragma unroll
  for (int j = 0; j < 8; ++j) {           // 8-channel chunks
    float av[8];
#pragma unroll
    for (int e = 0; e < 8; ++e)
      av[e] = x[((b * CCH + j * 8 + e) << 16) + p];     // residual (fp32)
#pragma unroll
    for (int t = 0; t < 9; ++t) {
      if (!ok[t]) continue;
      v4u32 vc = *(const v4u32*)(row + off[t] + 2 * CCH + j * 8);  // v section
#pragma unroll
      for (int e = 0; e < 4; ++e) {
        unsigned u = vc[e];
        av[2 * e]     += dots[t] * __uint_as_float(u << 16);
        av[2 * e + 1] += dots[t] * __uint_as_float(u & 0xFFFF0000u);
      }
    }
    v8u pack;
#pragma unroll
    for (int e = 0; e < 8; ++e) pack[e] = f2bf(av[e]);
    *(v8u*)(ypad + ybase + j * 8) = pack;  // b128 store
  }
}

// ---- kernel 3: 3x3 conv as 9 accumulated WMMA GEMMs, fused ReLU ----------
__global__ __launch_bounds__(256) void mlp_kernel(const u16* __restrict__ ypad,
                                                  const u16* __restrict__ wm,
                                                  float* __restrict__ out) {
  __shared__ u16 ys[3][130][CCH];   // halo tile: [row][pixel][channel], c-contig

  int tid = threadIdx.x;
  int blk = blockIdx.x;             // 2048 blocks = B * H * (W/128)
  int b   = blk >> 9;
  int rem = blk & 511;
  int h   = rem >> 1;
  int w0  = (rem & 1) << 7;

  // Tile = 3 contiguous lines of 130*64 bf16 at fixed stride in ypad
  // (zero-padding comes from the pre-zeroed halo; no masks needed).
  const u16* gsrc = ypad + ((size_t)((b * HP + h) * WP + (w0 + 1))) * CCH;
#if USE_TDM
  if (tid < 32) {
    // 2-D TDM: line = 130*64*2 B = 2080 x 8B units, 3 lines,
    // line stride = WP*64*2 B = 4160 x 8B units
    unsigned ldsa = (unsigned)(size_t)(void*)&ys[0][0][0];
    tdm_load(tdm_g0(ldsa, (unsigned long long)(size_t)(const void*)gsrc),
             tdm_g1(2080u, 3u, 2080u, 3u, 4160u));
    __builtin_amdgcn_s_wait_tensorcnt(0);
  }
#else
  {
    u16* ysf = &ys[0][0][0];
    for (int i = tid; i < 3 * 130 * CCH; i += 256) {
      int r = i / (130 * CCH);
      int o = i - r * (130 * CCH);
      ysf[i] = gsrc[(size_t)r * (WP * CCH) + o];
    }
  }
#endif
  __builtin_prefetch(wm, 0, 0);     // global_prefetch_b8 (weights -> L2/WGP$)
  __syncthreads();

  int lane = tid & 31;
  int wave = tid >> 5;
  int n    = lane & 15;
  int cb   = (lane < 16) ? 0 : 16;
  int klo  = (lane < 16) ? 0 : 8;
  int pxw  = wave * 16;

  v8f acc[4] = {};
#pragma unroll
  for (int t = 0; t < 9; ++t) {
    int dy   = t / 3;
    int slot = pxw + n + (t % 3);            // tile starts at global w0-1
#pragma unroll
    for (int ks = 0; ks < 2; ++ks) {
      int c0 = ks * 32 + cb;
      v16bf Bm = mk16(&ys[dy][slot][c0], &ys[dy][slot][c0 + 8]);
#pragma unroll
      for (int mt = 0; mt < 4; ++mt) {
        const u16* wr = wm + (size_t)(t * 64 + mt * 16 + n) * 64 + ks * 32;
        v16bf Am = mk16(wr + cb, wr + cb + 8);
        acc[mt] = wmma_bf16(Am, Bm, acc[mt]);
      }
    }
  }

  int mbase = (lane < 16) ? 0 : 8;
  int pout  = (h << 8) + w0 + pxw + n;
#pragma unroll
  for (int mt = 0; mt < 4; ++mt)
#pragma unroll
    for (int r = 0; r < 8; ++r) {
      int o = mt * 16 + mbase + r;
      out[((b * CCH + o) << 16) + pout] = fmaxf(acc[mt][r], 0.f);
    }
}

// ---- host launch ---------------------------------------------------------
extern "C" void kernel_launch(void* const* d_in, const int* in_sizes, int n_in,
                              void* d_out, int out_size, void* d_ws, size_t ws_size,
                              hipStream_t stream) {
  const float* x     = (const float*)d_in[0];
  const float* w_qkv = (const float*)d_in[1];
  const float* b_qkv = (const float*)d_in[2];
  const float* w_mlp = (const float*)d_in[3];
  float* out = (float*)d_out;

  // workspace: qkvp bf16 (96 MiB) | ypad bf16 (~32.8 MiB) | wm bf16 | wqb bf16
  u16* qkvp = (u16*)d_ws;
  u16* ypad = qkvp + (size_t)BATCH * PIX * OC3;
  u16* wm   = ypad + (size_t)BATCH * HP * WP * CCH;
  u16* wqb  = wm   + (size_t)9 * CCH * CCH;

  // Zero the halo (whole padded y buffer) — deterministic each call.
  hipMemsetAsync(ypad, 0, (size_t)BATCH * HP * WP * CCH * sizeof(u16), stream);

  conv_w_kernel<<<192,  256, 0, stream>>>(w_mlp, w_qkv, wm, wqb);
  qkv_kernel   <<<2048, 256, 0, stream>>>(x, wqb, b_qkv, qkvp);
  attn_kernel  <<<1024, 256, 0, stream>>>(qkvp, x, ypad);
  mlp_kernel   <<<2048, 256, 0, stream>>>(ypad, wm, out);
}
